// AttentionBlock_27745488732803
// MI455X (gfx1250) — matrix operand order repaired and checked by reading, not compiled, not run on hardware
//
#include <hip/hip_runtime.h>
#include <hip/hip_bf16.h>
#include <stdint.h>

// ---------------- types ----------------
typedef __attribute__((ext_vector_type(16))) __bf16 v16bf;
typedef __attribute__((ext_vector_type(8)))  __bf16 v8bf;
typedef __attribute__((ext_vector_type(8)))  float  v8f;
typedef __attribute__((ext_vector_type(4)))  unsigned int v4u;
typedef __attribute__((ext_vector_type(8)))  unsigned int v8u;

#define B_   16
#define C_   512
#define N_   1024
#define NH_  8
#define D_   64
#define O3_  1536

__device__ __forceinline__ v16bf ld_frag2(const __bf16* p0, const __bf16* p1) {
  v8bf a = *(const v8bf*)p0;
  v8bf b = *(const v8bf*)p1;
  v16bf r;
#pragma unroll
  for (int i = 0; i < 8; ++i) { r[i] = a[i]; r[8 + i] = b[i]; }
  return r;
}
// 16 contiguous bf16 (B operand: one column, 16 consecutive K)
__device__ __forceinline__ v16bf ld_frag16(const __bf16* p) { return ld_frag2(p, p + 8); }

__device__ __forceinline__ v8f wmma_bf16(v16bf a, v16bf b, v8f c) {
  return __builtin_amdgcn_wmma_f32_16x16x32_bf16(false, a, false, b, (short)0, c,
                                                 false, false);
}

// ---------------- Tensor Data Mover: 2D tile global -> LDS ------------------
// D# group0 (4 SGPRs) + group1 (8 SGPRs), bf16 elements (data_size=2B).
// tile_w elements along dim0, tile_h rows; tensor row stride = stride elems.
__device__ __forceinline__ void tdm_load_2d(const __bf16* lds_dst, const __bf16* gsrc,
                                            unsigned tile_w, unsigned tile_h,
                                            unsigned tensor_w, unsigned tensor_h,
                                            unsigned stride) {
  const unsigned lds_off = (unsigned)(uintptr_t)lds_dst;   // flat LDS addr low 32 = offset
  const unsigned long long ga = (unsigned long long)(uintptr_t)gsrc;
  v4u g0;
  g0[0] = 1u;                                            // count=1 (valid), user mode
  g0[1] = lds_off;                                       // lds_addr (bytes)
  g0[2] = (unsigned)ga;                                  // global_addr[31:0]
  g0[3] = ((unsigned)(ga >> 32) & 0x01FFFFFFu) | 0x80000000u; // addr[56:32] | type=2
  v8u g1;
  g1[0] = 0x10000u;                                      // data_size=1 -> 2 bytes
  g1[1] = (tensor_w & 0xFFFFu) << 16;                    // tensor_dim0[15:0]
  g1[2] = (tensor_w >> 16) | ((tensor_h & 0xFFFFu) << 16); // dim0[31:16] | dim1[15:0]
  g1[3] = (tensor_h >> 16) | (tile_w << 16);             // dim1[31:16] | tile_dim0
  g1[4] = tile_h;                                        // tile_dim1 ([15:0]), tile_dim2=0
  g1[5] = stride;                                        // tensor_dim0_stride[31:0]
  g1[6] = 0u;                                            // stride[47:32]=0
  g1[7] = 0u;
#pragma unroll
  for (int i = 0; i < 4; ++i) g0[i] = __builtin_amdgcn_readfirstlane(g0[i]);
#pragma unroll
  for (int i = 0; i < 8; ++i) g1[i] = __builtin_amdgcn_readfirstlane(g1[i]);
  asm volatile("tensor_load_to_lds %0, %1" :: "s"(g0), "s"(g1) : "memory");
}

// ---------------- 1) GroupNorm -> h[b][n][c] (bf16, token-major) ----------------
__global__ void gn_kernel(const float* __restrict__ x, const float* __restrict__ gamma,
                          const float* __restrict__ beta, __bf16* __restrict__ h) {
  __shared__ float s_sum[256];
  __shared__ float s_sq[256];
  const int b = blockIdx.x >> 5;
  const int g = blockIdx.x & 31;
  const float* xg = x + (((size_t)b * C_) + ((size_t)g << 4)) * N_;  // 16 x 1024
  float sum = 0.f, sq = 0.f;
  for (int i = threadIdx.x; i < 16384; i += 256) {
    float v = xg[i];
    sum += v; sq += v * v;
  }
  s_sum[threadIdx.x] = sum; s_sq[threadIdx.x] = sq;
  __syncthreads();
  for (int s = 128; s > 0; s >>= 1) {
    if ((int)threadIdx.x < s) {
      s_sum[threadIdx.x] += s_sum[threadIdx.x + s];
      s_sq[threadIdx.x]  += s_sq[threadIdx.x + s];
    }
    __syncthreads();
  }
  const float mean = s_sum[0] * (1.f / 16384.f);
  const float var  = s_sq[0] * (1.f / 16384.f) - mean * mean;
  const float rstd = rsqrtf(var + 1e-5f);
  for (int i = threadIdx.x; i < 16384; i += 256) {
    const int cl = i >> 10;
    const int n  = i & 1023;
    const int c  = (g << 4) + cl;
    float v = (xg[i] - mean) * rstd;
    v = v * gamma[c] + beta[c];
    h[(((size_t)b * N_ + n) * C_) + c] = (__bf16)v;
  }
}

// ---------------- 2) weights fp32 -> bf16 ----------------
__global__ void cvt_kernel(const float* __restrict__ wq, const float* __restrict__ wp,
                           __bf16* __restrict__ wqb, __bf16* __restrict__ wpb) {
  const int i = blockIdx.x * 256 + threadIdx.x;
  if (i < O3_ * C_) wqb[i] = (__bf16)wq[i];
  if (i < C_ * C_)  wpb[i] = (__bf16)wp[i];
}

// ---------------- 3) QKV GEMM (TDM-staged B tiles) ----------
// qkv[o][n] = sum_c w[o][c] h[n][c] + bias
// writes q_t[b][h][n][d] (scaled), k_t[b][h][n][d], v[b][h][d][n]
__global__ void qkv_kernel(const __bf16* __restrict__ h, const __bf16* __restrict__ w,
                           const float* __restrict__ bias,
                           __bf16* __restrict__ qt, __bf16* __restrict__ kt,
                           __bf16* __restrict__ vb) {
  __shared__ __bf16 htile[2][64 * 32];   // 64 n-rows x 32 c, double buffered
  const int wv = threadIdx.x >> 5;
  const int lane = threadIdx.x & 31;
  const int lo = lane & 15, hi = lane >> 4;
  const int o0 = blockIdx.x * 128 + wv * 16;
  const int n0 = blockIdx.y * 64;
  const int b  = blockIdx.z;

  const __bf16* hbase = h + (size_t)(b * N_ + n0) * C_;  // tile origin row
  if (wv == 0)
    tdm_load_2d(&htile[0][0], hbase, 32, 64, C_, N_, C_);

  v8f acc[4] = {};
  const __bf16* wrow = w + (size_t)(o0 + lo) * C_;
  for (int ks = 0; ks < 16; ++ks) {
    const int k0 = ks * 32;
    const int cur = ks & 1;
    if (wv == 0) {
      if (ks + 1 < 16) {
        tdm_load_2d(&htile[cur ^ 1][0], hbase + k0 + 32, 32, 64, C_, N_, C_);
        __builtin_amdgcn_s_wait_tensorcnt(1);
      } else {
        __builtin_amdgcn_s_wait_tensorcnt(0);
      }
    }
    __syncthreads();
    __builtin_prefetch(wrow + k0 + 128, 0, 1);
    v16bf af = ld_frag16(wrow + k0 + 16 * hi);
#pragma unroll
    for (int ns = 0; ns < 4; ++ns) {
      v16bf bf = ld_frag16(&htile[cur][(ns * 16 + lo) * 32 + 16 * hi]);
      acc[ns] = wmma_bf16(af, bf, acc[ns]);
    }
    __syncthreads();
  }

  const int s    = o0 >> 9;          // 0=q 1=k 2=v
  const int head = (o0 >> 6) & 7;
  const int dd0  = o0 & 63;
  const float scale = 0.125f;        // 64^-0.5 folded into q
#pragma unroll
  for (int ns = 0; ns < 4; ++ns) {
    const int n = n0 + ns * 16 + lo;
    if (s == 2) {
#pragma unroll
      for (int r = 0; r < 8; ++r) {
        const int dd = dd0 + 8 * hi + r;
        const float v = acc[ns][r] + bias[o0 + 8 * hi + r];
        vb[((size_t)((b * NH_ + head) * D_ + dd)) * N_ + n] = (__bf16)v;
      }
    } else {
      v8bf t;
#pragma unroll
      for (int r = 0; r < 8; ++r) {
        float v = acc[ns][r] + bias[o0 + 8 * hi + r];
        if (s == 0) v *= scale;
        t[r] = (__bf16)v;
      }
      __bf16* dst = (s == 0 ? qt : kt) +
                    ((size_t)((b * NH_ + head) * N_ + n)) * D_ + dd0 + 8 * hi;
      *(v8bf*)dst = t;
    }
  }
}

// ---------------- 4) flash attention (TDM-staged K/V tiles) ----------------
__global__ void attn_kernel(const __bf16* __restrict__ qt, const __bf16* __restrict__ kt,
                            const __bf16* __restrict__ vb, __bf16* __restrict__ ot) {
  __shared__ __bf16 kbuf[2][32 * 64];   // 32 m-rows x 64 d
  __shared__ __bf16 vbuf[2][64 * 32];   // 64 d-rows  x 32 m
  const int wv = threadIdx.x >> 5;
  const int lane = threadIdx.x & 31;
  const int lo = lane & 15, hi = lane >> 4;
  const int q0 = blockIdx.x * 128 + wv * 16;
  const int head = blockIdx.y, b = blockIdx.z;
  const size_t bh = (size_t)(b * NH_ + head);

  const __bf16* kt_bh = kt + bh * N_ * D_;
  const __bf16* vb_bh = vb + bh * D_ * N_;
  if (wv == 0) {
    tdm_load_2d(&kbuf[0][0], kt_bh, D_, 32, D_, N_, D_);
    tdm_load_2d(&vbuf[0][0], vb_bh, 32, D_, N_, D_, N_);
  }

  const __bf16* qrow = qt + (bh * N_ + q0 + lo) * D_;
  const v16bf qf0 = ld_frag16(qrow + 16 * hi);
  const v16bf qf1 = ld_frag16(qrow + 32 + 16 * hi);

  v8f oacc[4] = {};
  float m_run = -1e30f, l_run = 0.f;

  for (int it = 0; it < N_ / 32; ++it) {
    const int mt = it * 32;
    const int cur = it & 1;
    if (wv == 0) {
      if (it + 1 < N_ / 32) {
        tdm_load_2d(&kbuf[cur ^ 1][0], kt_bh + (size_t)(mt + 32) * D_, D_, 32, D_, N_, D_);
        tdm_load_2d(&vbuf[cur ^ 1][0], vb_bh + mt + 32, 32, D_, N_, D_, N_);
        __builtin_amdgcn_s_wait_tensorcnt(2);
      } else {
        __builtin_amdgcn_s_wait_tensorcnt(0);
      }
    }
    __syncthreads();

    v8f sc[2];
#pragma unroll
    for (int sub = 0; sub < 2; ++sub) {
      const __bf16* krow = &kbuf[cur][(sub * 16 + lo) * D_];
      v16bf a0 = ld_frag16(krow + 16 * hi);
      v16bf a1 = ld_frag16(krow + 32 + 16 * hi);
      v8f z = {0.f, 0.f, 0.f, 0.f, 0.f, 0.f, 0.f, 0.f};
      sc[sub] = wmma_bf16(a0, qf0, z);
      sc[sub] = wmma_bf16(a1, qf1, sc[sub]);
    }

    float tm = -1e30f;
#pragma unroll
    for (int r = 0; r < 8; ++r) { tm = fmaxf(tm, sc[0][r]); tm = fmaxf(tm, sc[1][r]); }
    tm = fmaxf(tm, __shfl_xor(tm, 16, 32));
    const float m_new = fmaxf(m_run, tm);
    const float resc = __expf(m_run - m_new);
#pragma unroll
    for (int dt = 0; dt < 4; ++dt)
#pragma unroll
      for (int r = 0; r < 8; ++r) oacc[dt][r] *= resc;
    l_run *= resc;

    float p0[8], p1[8], lsum = 0.f;
#pragma unroll
    for (int r = 0; r < 8; ++r) {
      p0[r] = __expf(sc[0][r] - m_new);
      p1[r] = __expf(sc[1][r] - m_new);
      lsum += p0[r] + p1[r];
    }
    lsum += __shfl_xor(lsum, 16, 32);
    l_run += lsum;
    m_run = m_new;

    // xor-16 exchange so each lane holds the 16 consecutive m values of its B-operand
    float recv[8];
#pragma unroll
    for (int r = 0; r < 8; ++r)
      recv[r] = __shfl_xor(hi ? p0[r] : p1[r], 16, 32);

    v16bf pf;
#pragma unroll
    for (int r = 0; r < 8; ++r) {
      if (hi == 0) { pf[r] = (__bf16)p0[r];   pf[8 + r] = (__bf16)recv[r]; }
      else         { pf[r] = (__bf16)recv[r]; pf[8 + r] = (__bf16)p1[r]; }
    }

#pragma unroll
    for (int dt = 0; dt < 4; ++dt) {
      const __bf16* vrow = &vbuf[cur][(dt * 16 + lo) * 32];
      v16bf a2 = ld_frag16(vrow + 16 * hi);
      oacc[dt] = wmma_bf16(a2, pf, oacc[dt]);
    }
    __syncthreads();
  }

  const float inv_l = 1.f / l_run;
  const int n = q0 + lo;
  __bf16* obase = ot + ((size_t)(b * N_ + n)) * C_ + head * D_;
#pragma unroll
  for (int dt = 0; dt < 4; ++dt) {
    v8bf t;
#pragma unroll
    for (int r = 0; r < 8; ++r) t[r] = (__bf16)(oacc[dt][r] * inv_l);
    *(v8bf*)(obase + dt * 16 + 8 * hi) = t;
  }
}

// ---------------- 5) proj GEMM + bias + residual (TDM-staged B tiles) -----------
__global__ void proj_kernel(const __bf16* __restrict__ ot, const __bf16* __restrict__ w,
                            const float* __restrict__ bias, const float* __restrict__ x,
                            float* __restrict__ out) {
  __shared__ __bf16 otile[2][64 * 32];
  const int wv = threadIdx.x >> 5;
  const int lane = threadIdx.x & 31;
  const int lo = lane & 15, hi = lane >> 4;
  const int o0 = blockIdx.x * 128 + wv * 16;
  const int n0 = blockIdx.y * 64;
  const int b  = blockIdx.z;

  const __bf16* obase = ot + (size_t)(b * N_ + n0) * C_;
  if (wv == 0)
    tdm_load_2d(&otile[0][0], obase, 32, 64, C_, N_, C_);

  v8f acc[4] = {};
  const __bf16* wrow = w + (size_t)(o0 + lo) * C_;
  for (int ks = 0; ks < 16; ++ks) {
    const int k0 = ks * 32;
    const int cur = ks & 1;
    if (wv == 0) {
      if (ks + 1 < 16) {
        tdm_load_2d(&otile[cur ^ 1][0], obase + k0 + 32, 32, 64, C_, N_, C_);
        __builtin_amdgcn_s_wait_tensorcnt(1);
      } else {
        __builtin_amdgcn_s_wait_tensorcnt(0);
      }
    }
    __syncthreads();
    v16bf af = ld_frag16(wrow + k0 + 16 * hi);
#pragma unroll
    for (int ns = 0; ns < 4; ++ns) {
      v16bf bf = ld_frag16(&otile[cur][(ns * 16 + lo) * 32 + 16 * hi]);
      acc[ns] = wmma_bf16(af, bf, acc[ns]);
    }
    __syncthreads();
  }
#pragma unroll
  for (int ns = 0; ns < 4; ++ns) {
    const int n = n0 + ns * 16 + lo;
#pragma unroll
    for (int r = 0; r < 8; ++r) {
      const int o = o0 + 8 * hi + r;
      const size_t idx = ((size_t)(b * C_ + o)) * N_ + n;
      out[idx] = x[idx] + acc[ns][r] + bias[o];
    }
  }
}

// ---------------- launch ----------------
extern "C" void kernel_launch(void* const* d_in, const int* in_sizes, int n_in,
                              void* d_out, int out_size, void* d_ws, size_t ws_size,
                              hipStream_t stream) {
  const float* x      = (const float*)d_in[0];
  const float* gamma  = (const float*)d_in[1];
  const float* beta   = (const float*)d_in[2];
  const float* w_qkv  = (const float*)d_in[3];
  const float* b_qkv  = (const float*)d_in[4];
  const float* w_proj = (const float*)d_in[5];
  const float* b_proj = (const float*)d_in[6];
  float* out = (float*)d_out;

  __bf16* ws = (__bf16*)d_ws;
  const size_t S = (size_t)B_ * N_ * C_;       // 8,388,608 elems
  __bf16* h    = ws;
  __bf16* qt   = h + S;
  __bf16* kt   = qt + S;
  __bf16* vb   = kt + S;
  __bf16* ot   = vb + S;
  __bf16* wqb  = ot + S;                       // 1536*512
  __bf16* wpb  = wqb + (size_t)O3_ * C_;       // 512*512

  gn_kernel<<<B_ * 32, 256, 0, stream>>>(x, gamma, beta, h);
  cvt_kernel<<<(O3_ * C_) / 256, 256, 0, stream>>>(w_qkv, w_proj, wqb, wpb);
  qkv_kernel<<<dim3(O3_ / 128, N_ / 64, B_), 256, 0, stream>>>(h, wqb, b_qkv, qt, kt, vb);
  attn_kernel<<<dim3(N_ / 128, NH_, B_), 256, 0, stream>>>(qt, kt, vb, ot);
  proj_kernel<<<dim3(C_ / 128, N_ / 64, B_), 256, 0, stream>>>(ot, wpb, b_proj, x, out);
}
